// GraspSamplerVAE_2680059593184
// MI455X (gfx1250) — compile-verified, matrix operand order checked by reading, not run on hardware
//
#include <hip/hip_runtime.h>
#include <hip/hip_bf16.h>

// GraspSamplerVAE forward for MI455X (gfx1250, wave32).
// Strategy: all shared-MLP / FC layers run as WMMA f16->f32 GEMMs
// (v_wmma_f32_16x16x32_f16), activations stored f16 (bandwidth roofline:
// 23.3 TB/s HBM dominates; compute is trivial). BN stats are a deterministic
// per-channel reduction; BN+ReLU (+max over neighborhood) fused into epilogue
// kernels. Geometry (FPS / ball query) computed once, shared by enc+dec.

#define BATCH 128
#define NPTS  2048

typedef __attribute__((ext_vector_type(16))) _Float16     v16h;
typedef __attribute__((ext_vector_type(8)))  float        v8f;
typedef __attribute__((ext_vector_type(4)))  unsigned int u32x4;

// ---------------------------------------------------------------- utilities
__global__ void k_cast16(const float* __restrict__ a, _Float16* __restrict__ o, int n) {
  int t = blockIdx.x * 256 + threadIdx.x;
  if (t < n) o[t] = (_Float16)a[t];
}

__global__ void k_prepw(const float* __restrict__ W, _Float16* __restrict__ W16,
                        int N, int cin, int Kp) {
  int t = blockIdx.x * 256 + threadIdx.x;
  if (t >= N * Kp) return;
  int nrow = t / Kp, k = t % Kp;
  W16[t] = (k < cin) ? (_Float16)W[nrow * cin + k] : (_Float16)0.f;
}

// ---------------------------------------------------------------- WMMA GEMM
// Y[M,N] = X[M,Kp] * W16[N,Kp]^T + bias ; f16 in, f32 accumulate, f16 out.
// One wave -> one 16x16 tile. Block = 8 waves covering 128 rows x 16 cols.
__global__ __launch_bounds__(256)
void k_gemm(const _Float16* __restrict__ X, const _Float16* __restrict__ W16,
            const float* __restrict__ bias, _Float16* __restrict__ Y,
            int M, int N, int Kp) {
  const int wave = threadIdx.x >> 5;
  const int lane = threadIdx.x & 31;
  const int m0 = blockIdx.y * 128 + wave * 16;
  const int n0 = blockIdx.x * 16;
  if (m0 >= M) return;                       // wave-uniform
  const int nl = lane & 15;
  const int hi = lane >> 4;
  const _Float16* xrow = X   + (size_t)(m0 + nl) * Kp;
  const _Float16* wrow = W16 + (size_t)(n0 + nl) * Kp;
  v8f acc = {};
  for (int k0 = 0; k0 < Kp; k0 += 32) {
    union { u32x4 u[2]; v16h h; } a, b;
    // 16-bit A 16x32 layout: lane group selects K subset {hi*8..} / {16+hi*8..}
    a.u[0] = *(const u32x4*)(xrow + k0 + hi * 8);
    a.u[1] = *(const u32x4*)(xrow + k0 + 16 + hi * 8);
    b.u[0] = *(const u32x4*)(wrow + k0 + hi * 8);
    b.u[1] = *(const u32x4*)(wrow + k0 + 16 + hi * 8);
    acc = __builtin_amdgcn_wmma_f32_16x16x32_f16(false, a.h, false, b.h,
                                                 (short)0, acc, false, false);
  }
  const float bv = bias[n0 + nl];
#pragma unroll
  for (int r = 0; r < 8; r++) {
    // C/D layout: VGPR r -> row m0 + r + 8*hi, col n0 + (lane&15)
    Y[(size_t)(m0 + r + 8 * hi) * N + (n0 + nl)] = (_Float16)(acc[r] + bv);
  }
}

// -------------------------------------------------- per-channel BN statistics
__global__ void k_stats(const _Float16* __restrict__ Y, float* __restrict__ s_sum,
                        float* __restrict__ s_sq, long long M, int N) {
  int n = blockIdx.x;
  float s = 0.f, q = 0.f;
  for (long long r = threadIdx.x; r < M; r += 256) {
    float v = (float)Y[(size_t)r * N + n];
    s += v; q += v * v;
  }
  __shared__ float sh[256], shq[256];
  sh[threadIdx.x] = s; shq[threadIdx.x] = q;
  __syncthreads();
  for (int st = 128; st > 0; st >>= 1) {
    if (threadIdx.x < st) { sh[threadIdx.x] += sh[threadIdx.x + st];
                            shq[threadIdx.x] += shq[threadIdx.x + st]; }
    __syncthreads();
  }
  if (threadIdx.x == 0) { s_sum[n] = sh[0]; s_sq[n] = shq[0]; }
}

__global__ void k_bnrelu(_Float16* Y, const float* __restrict__ s_sum,
                         const float* __restrict__ s_sq,
                         const float* __restrict__ gamma, const float* __restrict__ beta,
                         float invM, long long total, int N) {
  long long t = (long long)blockIdx.x * 256 + threadIdx.x;
  if (t >= total) return;
  int n = (int)(t % N);
  float m = s_sum[n] * invM;
  float var = s_sq[n] * invM - m * m;
  float rstd = rsqrtf(var + 1e-5f);
  float v = ((float)Y[t] - m) * rstd * gamma[n] + beta[n];
  Y[t] = (_Float16)fmaxf(v, 0.f);
}

// BN + ReLU + max over the neighborhood (k) dimension; output f32 features.
__global__ void k_bnrelu_maxk(const _Float16* __restrict__ Y,
                              const float* __restrict__ s_sum, const float* __restrict__ s_sq,
                              const float* __restrict__ gamma, const float* __restrict__ beta,
                              float invM, float* __restrict__ F,
                              int groups, int Kn, int N) {
  int t = blockIdx.x * 256 + threadIdx.x;
  if (t >= groups * N) return;
  int g = t / N, n = t % N;
  float m = s_sum[n] * invM;
  float var = s_sq[n] * invM - m * m;
  float rstd = rsqrtf(var + 1e-5f);
  float ga = gamma[n], be = beta[n];
  float best = -3.4e38f;
  const _Float16* p = Y + (size_t)g * Kn * N + n;
  for (int k = 0; k < Kn; k++) {
    float v = ((float)p[(size_t)k * N] - m) * rstd * ga + be;
    best = fmaxf(best, v);
  }
  F[(size_t)g * N + n] = fmaxf(best, 0.f);   // relu(max) == max(relu)
}

// ---------------------------------------------------------------- geometry
// Farthest point sampling; one block per batch. Matches lax.scan semantics:
// outputs the incoming 'far' each step, starting at 0.
__global__ __launch_bounds__(256)
void k_fps(const float* __restrict__ xyz, int* __restrict__ fidx,
           float* __restrict__ new_xyz, int n, int np) {
  __shared__ float sx[NPTS], sy[NPTS], sz[NPTS], sd[NPTS];
  __shared__ float rv[256];
  __shared__ int   ri[256];
  __shared__ int   s_far;
  int b = blockIdx.x, tid = threadIdx.x;
  for (int j = tid; j < n; j += 256) {
    sx[j] = xyz[((size_t)b * n + j) * 3 + 0];
    sy[j] = xyz[((size_t)b * n + j) * 3 + 1];
    sz[j] = xyz[((size_t)b * n + j) * 3 + 2];
    sd[j] = 1e10f;
  }
  if (tid == 0) s_far = 0;
  __syncthreads();
  for (int i = 0; i < np; i++) {
    int far = s_far;
    float cx = sx[far], cy = sy[far], cz = sz[far];
    if (tid == 0) {
      fidx[b * np + i] = far;
      new_xyz[((size_t)b * np + i) * 3 + 0] = cx;
      new_xyz[((size_t)b * np + i) * 3 + 1] = cy;
      new_xyz[((size_t)b * np + i) * 3 + 2] = cz;
    }
    float bv = -1.f; int bi = 0;
    for (int j = tid; j < n; j += 256) {
      float dx = sx[j] - cx, dy = sy[j] - cy, dz = sz[j] - cz;
      float nd = fminf(sd[j], dx * dx + dy * dy + dz * dz);
      sd[j] = nd;
      if (nd > bv) { bv = nd; bi = j; }      // first-occurrence argmax
    }
    rv[tid] = bv; ri[tid] = bi;
    __syncthreads();
    for (int st = 128; st > 0; st >>= 1) {
      if (tid < st) {
        if (rv[tid + st] > rv[tid] ||
            (rv[tid + st] == rv[tid] && ri[tid + st] < ri[tid])) {
          rv[tid] = rv[tid + st]; ri[tid] = ri[tid + st];
        }
      }
      __syncthreads();
    }
    if (tid == 0) s_far = ri[0];
    __syncthreads();
  }
}

// Ball query: ascending valid indices, padded with first (or 0 if none).
__global__ void k_ball(const float* __restrict__ xyz, const float* __restrict__ new_xyz,
                       int* __restrict__ idx, int S, int n, int ns, float r2) {
  int t = blockIdx.x * 256 + threadIdx.x;
  if (t >= BATCH * S) return;
  int b = t / S;
  const float* q = new_xyz + (size_t)t * 3;
  const float* P = xyz + (size_t)b * n * 3;
  int* row = idx + (size_t)t * ns;
  int cnt = 0;
  for (int j = 0; j < n && cnt < ns; j++) {
    float dx = P[j * 3 + 0] - q[0];
    float dy = P[j * 3 + 1] - q[1];
    float dz = P[j * 3 + 2] - q[2];
    if (dx * dx + dy * dy + dz * dz < r2) row[cnt++] = j;
  }
  int first = (cnt > 0) ? row[0] : 0;
  for (int k = cnt; k < ns; k++) row[k] = first;
}

// Build grouped input rows: [g_xyz - new_xyz (3), feats (Cf)], f16, zero-pad Kp.
__global__ void k_group(const float* __restrict__ xyz, const float* __restrict__ new_xyz,
                        const float* __restrict__ feats, const int* __restrict__ idx,
                        _Float16* __restrict__ X, int S, int Kn, int n, int Cf, int Kp) {
  long long t = (long long)blockIdx.x * 256 + threadIdx.x;
  long long total = (long long)BATCH * S * Kn;
  if (t >= total) return;
  int b = (int)(t / ((long long)S * Kn));
  int s = (int)((t / Kn) % S);
  int j = idx[t];
  const float* p = xyz + ((size_t)b * n + j) * 3;
  const float* q = new_xyz + ((size_t)b * S + s) * 3;
  _Float16* o = X + (size_t)t * Kp;
  o[0] = (_Float16)(p[0] - q[0]);
  o[1] = (_Float16)(p[1] - q[1]);
  o[2] = (_Float16)(p[2] - q[2]);
  const float* f = feats + ((size_t)b * n + j) * Cf;
  for (int c = 0; c < Cf; c++) o[3 + c] = (_Float16)f[c];
  for (int k = 3 + Cf; k < Kp; k++) o[k] = (_Float16)0.f;
}

// _sa_all input rows: [xyz (3), feats (Cf)], f16, zero-pad Kp.
__global__ void k_group_all(const float* __restrict__ xyz, const float* __restrict__ feats,
                            _Float16* __restrict__ X, int S, int Cf, int Kp) {
  int t = blockIdx.x * 256 + threadIdx.x;
  if (t >= BATCH * S) return;
  const float* p = xyz + (size_t)t * 3;
  const float* f = feats + (size_t)t * Cf;
  _Float16* o = X + (size_t)t * Kp;
  o[0] = (_Float16)p[0]; o[1] = (_Float16)p[1]; o[2] = (_Float16)p[2];
  for (int c = 0; c < Cf; c++) o[3 + c] = (_Float16)f[c];
  for (int k = 3 + Cf; k < Kp; k++) o[k] = (_Float16)0.f;
}

// Per-point feature tensors for encoder/decoder stage-1 grouping.
__global__ void k_feats_enc(const float* __restrict__ pc, const float* __restrict__ grasp,
                            float* __restrict__ f) {
  int t = blockIdx.x * 256 + threadIdx.x;
  if (t >= BATCH * NPTS) return;
  int b = t / NPTS;
  const float* p = pc + (size_t)t * 3;
  float* o = f + (size_t)t * 19;
  o[0] = p[0]; o[1] = p[1]; o[2] = p[2];
  const float* g = grasp + b * 16;
  for (int c = 0; c < 16; c++) o[3 + c] = g[c];
}

__global__ void k_feats_dec(const float* __restrict__ pc, const float* __restrict__ z,
                            float* __restrict__ f) {
  int t = blockIdx.x * 256 + threadIdx.x;
  if (t >= BATCH * NPTS) return;
  int b = t / NPTS;
  const float* p = pc + (size_t)t * 3;
  float* o = f + (size_t)t * 5;
  o[0] = p[0]; o[1] = p[1]; o[2] = p[2];
  o[3] = z[b * 2 + 0]; o[4] = z[b * 2 + 1];
}

// ---------------------------------------------------------------- tiny heads
__global__ void k_latent(const _Float16* __restrict__ h,
                         const float* __restrict__ Wmu, const float* __restrict__ bmu,
                         const float* __restrict__ Wlv, const float* __restrict__ blv,
                         const float* __restrict__ eps, float* __restrict__ z,
                         float* __restrict__ out) {
  int b = blockIdx.x * 64 + threadIdx.x;
  if (b >= BATCH) return;
  const _Float16* hr = h + (size_t)b * 1024;
  for (int o = 0; o < 2; o++) {
    float mu = bmu[o], lv = blv[o];
    const float* wm = Wmu + o * 1024;
    const float* wl = Wlv + o * 1024;
    for (int k = 0; k < 1024; k++) {
      float hv = (float)hr[k];
      mu += hv * wm[k];
      lv += hv * wl[k];
    }
    out[1024 + b * 2 + o] = mu;       // mu after qt(896)+conf(128)
    out[1280 + b * 2 + o] = lv;       // logvar
    z[b * 2 + o] = mu + eps[b * 2 + o] * expf(0.5f * lv);
  }
}

__global__ void k_heads(const _Float16* __restrict__ x,
                        const float* __restrict__ Wq, const float* __restrict__ bq,
                        const float* __restrict__ Wt, const float* __restrict__ bt,
                        const float* __restrict__ Wc, const float* __restrict__ bc,
                        float* __restrict__ out) {
  int b = blockIdx.x * 64 + threadIdx.x;
  if (b >= BATCH) return;
  const _Float16* xr = x + (size_t)b * 1024;
  float q[4], tt[3], cf;
  for (int o = 0; o < 4; o++) {
    float a = bq[o]; const float* w = Wq + o * 1024;
    for (int k = 0; k < 1024; k++) a += (float)xr[k] * w[k];
    q[o] = a;
  }
  for (int o = 0; o < 3; o++) {
    float a = bt[o]; const float* w = Wt + o * 1024;
    for (int k = 0; k < 1024; k++) a += (float)xr[k] * w[k];
    tt[o] = a;
  }
  { float a = bc[0]; for (int k = 0; k < 1024; k++) a += (float)xr[k] * Wc[k]; cf = a; }
  float nq = sqrtf(q[0] * q[0] + q[1] * q[1] + q[2] * q[2] + q[3] * q[3]);
  float inv = 1.f / fmaxf(nq, 1e-12f);
  float* qt = out + b * 7;
  qt[0] = q[0] * inv; qt[1] = q[1] * inv; qt[2] = q[2] * inv; qt[3] = q[3] * inv;
  qt[4] = tt[0]; qt[5] = tt[1]; qt[6] = tt[2];
  out[896 + b] = 1.f / (1.f + expf(-cf));
}

// ================================================================ host side
struct LD { int iw, ib, ig, ibe; int cin, Kp, N; };

extern "C" void kernel_launch(void* const* d_in, const int* in_sizes, int n_in,
                              void* d_out, int out_size, void* d_ws, size_t ws_size,
                              hipStream_t stream) {
  (void)in_sizes; (void)n_in; (void)out_size; (void)ws_size;
  const float* pc    = (const float*)d_in[0];
  const float* grasp = (const float*)d_in[1];
  const float* eps   = (const float*)d_in[2];
  float* out = (float*)d_out;
  auto P = [&](int i) -> const float* { return (const float*)d_in[i]; };

  // Params flattened in setup_inputs() insertion order; each MLP layer = W,b,gamma,beta.
  static const LD enc_sa1[3] = {{3,4,5,6, 22,32,64},    {7,8,9,10, 64,64,64},    {11,12,13,14, 64,64,128}};
  static const LD enc_sa2[3] = {{15,16,17,18, 131,160,128},{19,20,21,22, 128,128,128},{23,24,25,26, 128,128,256}};
  static const LD enc_sa3[3] = {{27,28,29,30, 259,288,256},{31,32,33,34, 256,256,256},{35,36,37,38, 256,256,512}};
  static const LD enc_fc[2]  = {{39,40,41,42, 512,512,1024},{43,44,45,46, 1024,1024,1024}};
  static const LD dec_sa1[3] = {{51,52,53,54, 8,32,64},  {55,56,57,58, 64,64,64},  {59,60,61,62, 64,64,128}};
  static const LD dec_sa2[3] = {{63,64,65,66, 131,160,128},{67,68,69,70, 128,128,128},{71,72,73,74, 128,128,256}};
  static const LD dec_sa3[3] = {{75,76,77,78, 259,288,256},{79,80,81,82, 256,256,256},{83,84,85,86, 256,256,512}};
  static const LD dec_fc[2]  = {{87,88,89,90, 512,512,1024},{91,92,93,94, 1024,1024,1024}};
  const LD* G[8] = {enc_sa1, enc_sa2, enc_sa3, enc_fc, dec_sa1, dec_sa2, dec_sa3, dec_fc};
  const int GC[8] = {3, 3, 3, 2, 3, 3, 3, 2};

  // ---- workspace bump allocator
  char* base = (char*)d_ws; size_t off = 0;
  auto alloc = [&](size_t bytes) -> void* {
    void* p = base + off; off = (off + bytes + 255) & ~(size_t)255; return p;
  };

  // f16 padded weights for all 22 WMMA layers
  _Float16* WP[8][3];
  for (int g = 0; g < 8; g++)
    for (int l = 0; l < GC[g]; l++) {
      const LD& L = G[g][l];
      WP[g][l] = (_Float16*)alloc((size_t)L.N * L.Kp * 2);
      int tot = L.N * L.Kp;
      k_prepw<<<(tot + 255) / 256, 256, 0, stream>>>(P(L.iw), WP[g][l], L.N, L.cin, L.Kp);
    }

  float* stats = (float*)alloc(2048 * 4);  // [sum(1024) | sumsq(1024)]

  int*   fidx1 = (int*)alloc((size_t)BATCH * 128 * 4);
  float* nx1   = (float*)alloc((size_t)BATCH * 128 * 3 * 4);
  int*   idx1  = (int*)alloc((size_t)BATCH * 128 * 64 * 4);
  int*   fidx2 = (int*)alloc((size_t)BATCH * 32 * 4);
  float* nx2   = (float*)alloc((size_t)BATCH * 32 * 3 * 4);
  int*   idx2  = (int*)alloc((size_t)BATCH * 32 * 128 * 4);

  float* f0e  = (float*)alloc((size_t)BATCH * NPTS * 19 * 4);
  float* f0d  = (float*)alloc((size_t)BATCH * NPTS * 5 * 4);
  float* fsa1 = (float*)alloc((size_t)BATCH * 128 * 128 * 4);
  float* fsa2 = (float*)alloc((size_t)BATCH * 32 * 256 * 4);
  float* h512 = (float*)alloc((size_t)BATCH * 512 * 4);
  _Float16* h16 = (_Float16*)alloc((size_t)BATCH * 512 * 2);
  _Float16* fcA = (_Float16*)alloc((size_t)BATCH * 1024 * 2);
  _Float16* fcB = (_Float16*)alloc((size_t)BATCH * 1024 * 2);
  float* zbuf = (float*)alloc((size_t)BATCH * 2 * 4);

  // ping-pong activation buffers (f16); sized for SA2 X0 (160MiB) / SA L2 out (256MiB)
  _Float16* r0h = (_Float16*)alloc((size_t)167772160);
  _Float16* r1h = (_Float16*)alloc((size_t)268435456);

  // ---- GEMM + BN helpers
  auto rungemm = [&](const _Float16* Xb, _Float16* Yb, const LD& L,
                     const _Float16* Wp, int M) {
    dim3 grd((unsigned)(L.N / 16), (unsigned)(M / 128));
    k_gemm<<<grd, 256, 0, stream>>>(Xb, Wp, P(L.ib), Yb, M, L.N, L.Kp);
    k_stats<<<L.N, 256, 0, stream>>>(Yb, stats, stats + 1024, (long long)M, L.N);
  };
  auto runbn = [&](_Float16* Yb, const LD& L, int M) {
    long long total = (long long)M * L.N;
    k_bnrelu<<<(unsigned)((total + 255) / 256), 256, 0, stream>>>(
        Yb, stats, stats + 1024, P(L.ig), P(L.ibe), 1.f / (float)M, total, L.N);
  };
  auto runmlp = [&](int g, int M, float* F, int groupsN, int Kn) {
    _Float16* cur = r0h; _Float16* nxt = r1h;
    for (int l = 0; l < GC[g]; l++) {
      const LD& L = G[g][l];
      rungemm(cur, nxt, L, WP[g][l], M);
      if (l < GC[g] - 1) {
        runbn(nxt, L, M);
        _Float16* t = cur; cur = nxt; nxt = t;
      } else {
        int tot = groupsN * L.N;
        k_bnrelu_maxk<<<(tot + 255) / 256, 256, 0, stream>>>(
            nxt, stats, stats + 1024, P(L.ig), P(L.ibe), 1.f / (float)M,
            F, groupsN, Kn, L.N);
      }
    }
  };
  auto runfc = [&](int g, const _Float16* Xin) {
    rungemm(Xin, fcA, G[g][0], WP[g][0], BATCH); runbn(fcA, G[g][0], BATCH);
    rungemm(fcA, fcB, G[g][1], WP[g][1], BATCH); runbn(fcB, G[g][1], BATCH);
  };

  // ---- geometry (shared by encoder and decoder: FPS/ball depend only on pc)
  k_fps<<<BATCH, 256, 0, stream>>>(pc, fidx1, nx1, NPTS, 128);
  k_ball<<<(BATCH * 128 + 255) / 256, 256, 0, stream>>>(pc, nx1, idx1, 128, NPTS, 64, 0.02f * 0.02f);
  k_fps<<<BATCH, 256, 0, stream>>>(nx1, fidx2, nx2, 128, 32);
  k_ball<<<(BATCH * 32 + 255) / 256, 256, 0, stream>>>(nx1, nx2, idx2, 32, 128, 128, 0.04f * 0.04f);

  const int M1 = BATCH * 128 * 64;   // 1,048,576
  const int M2 = BATCH * 32 * 128;   //   524,288
  const int M3 = BATCH * 32;         //     4,096

  // ---- encoder
  k_feats_enc<<<(BATCH * NPTS + 255) / 256, 256, 0, stream>>>(pc, grasp, f0e);
  k_group<<<(unsigned)((M1 + 255) / 256), 256, 0, stream>>>(pc, nx1, f0e, idx1, r0h, 128, 64, NPTS, 19, 32);
  runmlp(0, M1, fsa1, BATCH * 128, 64);
  k_group<<<(unsigned)((M2 + 255) / 256), 256, 0, stream>>>(nx1, nx2, fsa1, idx2, r0h, 32, 128, 128, 128, 160);
  runmlp(1, M2, fsa2, BATCH * 32, 128);
  k_group_all<<<(M3 + 255) / 256, 256, 0, stream>>>(nx2, fsa2, r0h, 32, 256, 288);
  runmlp(2, M3, h512, BATCH, 32);
  k_cast16<<<(BATCH * 512 + 255) / 256, 256, 0, stream>>>(h512, h16, BATCH * 512);
  runfc(3, h16);
  k_latent<<<2, 64, 0, stream>>>(fcB, P(47), P(48), P(49), P(50), eps, zbuf, out);

  // ---- decoder (reuses geometry + feature buffers)
  k_feats_dec<<<(BATCH * NPTS + 255) / 256, 256, 0, stream>>>(pc, zbuf, f0d);
  k_group<<<(unsigned)((M1 + 255) / 256), 256, 0, stream>>>(pc, nx1, f0d, idx1, r0h, 128, 64, NPTS, 5, 32);
  runmlp(4, M1, fsa1, BATCH * 128, 64);
  k_group<<<(unsigned)((M2 + 255) / 256), 256, 0, stream>>>(nx1, nx2, fsa1, idx2, r0h, 32, 128, 128, 128, 160);
  runmlp(5, M2, fsa2, BATCH * 32, 128);
  k_group_all<<<(M3 + 255) / 256, 256, 0, stream>>>(nx2, fsa2, r0h, 32, 256, 288);
  runmlp(6, M3, h512, BATCH, 32);
  k_cast16<<<(BATCH * 512 + 255) / 256, 256, 0, stream>>>(h512, h16, BATCH * 512);
  runfc(7, h16);
  k_heads<<<2, 64, 0, stream>>>(fcB, P(95), P(96), P(97), P(98), P(99), P(100), out);
}
